// Regressor_17841294148217
// MI455X (gfx1250) — matrix operand order repaired and checked by reading, not compiled, not run on hardware
//
#include <hip/hip_runtime.h>

// ---------------------------------------------------------------------------
// Regressor pipeline for MI455X (gfx1250, wave32, WMMA).
//   autoencoder convs -> WMMA f16 implicit GEMM (16x16x32, f32 accum)
//   CCL -> LDS-tiled iterative 3x3 min-label propagation
//   extract/match/MLP -> small dedicated kernels
// ---------------------------------------------------------------------------

#define HH 384
#define WW 384
#define SENT 147456           // H*W sentinel
#define IMGS 16               // images per chunk (4 chunks: x1 lo/hi, x2 lo/hi)

typedef __attribute__((ext_vector_type(16))) _Float16 v16h;
typedef __attribute__((ext_vector_type(8)))  float    v8f;

__device__ __forceinline__ float frelu(float x) { return x > 0.f ? x : 0.f; }
__device__ __forceinline__ int   imin2(int a, int b) { return a < b ? a : b; }

// ------------------------- weight packing ----------------------------------
// forward conv weights (O,I,4,4) flatten exactly to Bt[n][k=i*16+ky*4+kx]
__global__ void pack_conv_k(const float* __restrict__ w, _Float16* __restrict__ Bt, int total) {
  int t = blockIdx.x * blockDim.x + threadIdx.x;
  if (t < total) Bt[t] = (_Float16)w[t];
}

// convT weights (I,O,4,4) -> per-parity Bt4[p][n][k=i*4 + ty*2 + tx]
__global__ void pack_convT_k(const float* __restrict__ w, _Float16* __restrict__ Bt4,
                             int inC, int outC) {
  int K = inC * 4;
  int total = 4 * outC * K;
  int t = blockIdx.x * blockDim.x + threadIdx.x;
  if (t >= total) return;
  int p = t / (outC * K);
  int r = t % (outC * K);
  int n = r / K, k = r % K;
  int i = k >> 2, tt = k & 3, ty = tt >> 1, tx = tt & 1;
  int po = p >> 1, px = p & 1;
  int ky = 2 * ty + 1 - po;
  int kx = 2 * tx + 1 - px;
  Bt4[t] = (_Float16)w[((i * outC + n) * 4 + ky) * 4 + kx];
}

// ------------------------- e1: conv 1->16, 384->192 (direct f32) -----------
__global__ void conv_e1_k(const float* __restrict__ x, const float* __restrict__ w,
                          const float* __restrict__ b, _Float16* __restrict__ d2cat) {
  int img = blockIdx.z, o = blockIdx.y;
  int t = blockIdx.x * 256 + threadIdx.x;      // < 192*192
  int oy = t / 192, ox = t % 192;
  float s = b[o];
#pragma unroll
  for (int ky = 0; ky < 4; ++ky) {
    int iy = 2 * oy - 1 + ky;
    if ((unsigned)iy >= (unsigned)HH) continue;
#pragma unroll
    for (int kx = 0; kx < 4; ++kx) {
      int ix = 2 * ox - 1 + kx;
      if ((unsigned)ix >= (unsigned)WW) continue;
      s += x[(size_t)img * (HH * WW) + iy * WW + ix] * w[o * 16 + ky * 4 + kx];
    }
  }
  // skip-concat slot: channels 16..31 of d2cat
  d2cat[((size_t)img * 32 + 16 + o) * 36864 + t] = (_Float16)frelu(s);
}

// ------------------------- forward conv via WMMA ---------------------------
// grid (outW/16, outH, img), block 32*(N/16). M-tile = 16 consecutive ox.
__global__ void __launch_bounds__(128)
conv_fwd_wmma(const _Float16* __restrict__ in, int inImgStride,
              int inC, int inH, int inW,
              const _Float16* __restrict__ Bt, int K,
              const float* __restrict__ bias,
              _Float16* __restrict__ out, int outImgStride, int chBase,
              int N, int outH, int outW) {
  const int tid = threadIdx.x, lane = tid & 31, wave = tid >> 5;
  const int row = lane & 15, hi = lane >> 4;
  const int oxBase = blockIdx.x * 16;
  const int oy = blockIdx.y, img = blockIdx.z;
  const _Float16* inImg = in + (size_t)img * inImgStride;
  __shared__ _Float16 As[16][40];
  v8f acc = {};
  const int n0 = wave * 16, col = lane & 15;
  (void)inC;
  for (int kc = 0; kc < K; kc += 32) {
    for (int e = tid; e < 512; e += blockDim.x) {   // stage A tile 16x32
      int r = e >> 5, kk = e & 31, k = kc + kk;
      int ci = k >> 4, rem = k & 15, ky = rem >> 2, kx = rem & 3;
      int ox = oxBase + r;
      int iy = 2 * oy - 1 + ky, ix = 2 * ox - 1 + kx;
      _Float16 v = (_Float16)0.f;
      if ((unsigned)iy < (unsigned)inH && (unsigned)ix < (unsigned)inW)
        v = inImg[((size_t)ci * inH + iy) * inW + ix];
      As[r][kk] = v;
    }
    __syncthreads();
    v16h a, b;
    int b1 = hi * 16;
#pragma unroll
    for (int h = 0; h < 16; ++h) a[h] = As[row][b1 + h];
    const _Float16* bp = Bt + (size_t)(n0 + col) * K + kc + hi * 16;
#pragma unroll
    for (int h = 0; h < 16; ++h) b[h] = bp[h];
    acc = __builtin_amdgcn_wmma_f32_16x16x32_f16(false, a, false, b, (short)0, acc, false, false);
    __syncthreads();
  }
  int n = n0 + col;
  float bs = bias[n];
  size_t ob = (size_t)img * outImgStride + ((size_t)(chBase + n) * outH + oy) * outW + oxBase;
#pragma unroll
  for (int f = 0; f < 8; ++f) {
    int m = f + 8 * hi;
    out[ob + m] = (_Float16)frelu(acc[f] + bs);
  }
}

// ------------------------- transposed conv via WMMA ------------------------
// grid.x encodes (tx, px): ox = tx*32 + px + 2*row. K = inC*4 (exact taps).
__global__ void __launch_bounds__(128)
convT_wmma(const _Float16* __restrict__ in, int inImgStride,
           int inC, int inH, int inW,
           const _Float16* __restrict__ Bt4, int K,
           const float* __restrict__ bias,
           _Float16* __restrict__ out, int outImgStride, int chBase,
           int N, int outH, int outW) {
  const int tid = threadIdx.x, lane = tid & 31, wave = tid >> 5;
  const int row = lane & 15, hi = lane >> 4;
  const int px = blockIdx.x & 1, tx = blockIdx.x >> 1;
  const int ox0 = tx * 32 + px;
  const int oy = blockIdx.y, po = oy & 1;
  const int img = blockIdx.z;
  const _Float16* inImg = in + (size_t)img * inImgStride;
  const _Float16* Bt = Bt4 + (size_t)(po * 2 + px) * N * K;
  __shared__ _Float16 As[16][40];
  v8f acc = {};
  const int n0 = wave * 16, col = lane & 15;
  const int iyb = (oy + po) >> 1;
  (void)inC;
  for (int kc = 0; kc < K; kc += 32) {
    for (int e = tid; e < 512; e += blockDim.x) {
      int r = e >> 5, kk = e & 31, k = kc + kk;
      int ci = k >> 2, t = k & 3, ty = t >> 1, txk = t & 1;
      int iy = iyb - ty;
      int ix = tx * 16 + px + r - txk;
      _Float16 v = (_Float16)0.f;
      if ((unsigned)iy < (unsigned)inH && (unsigned)ix < (unsigned)inW)
        v = inImg[((size_t)ci * inH + iy) * inW + ix];
      As[r][kk] = v;
    }
    __syncthreads();
    v16h a, b;
    int b1 = hi * 16;
#pragma unroll
    for (int h = 0; h < 16; ++h) a[h] = As[row][b1 + h];
    const _Float16* bp = Bt + (size_t)(n0 + col) * K + kc + hi * 16;
#pragma unroll
    for (int h = 0; h < 16; ++h) b[h] = bp[h];
    acc = __builtin_amdgcn_wmma_f32_16x16x32_f16(false, a, false, b, (short)0, acc, false, false);
    __syncthreads();
  }
  int n = n0 + col;
  float bs = bias[n];
  size_t ob = (size_t)img * outImgStride + ((size_t)(chBase + n) * outH + oy) * outW + ox0;
#pragma unroll
  for (int f = 0; f < 8; ++f) {
    int m = f + 8 * hi;
    out[ob + 2 * m] = (_Float16)frelu(acc[f] + bs);
  }
}

// ------------------------- d1: convT 32->1 + sigmoid (direct) --------------
__global__ void d1_sigmoid_k(const _Float16* __restrict__ d2cat, const float* __restrict__ w,
                             const float* __restrict__ b, float* __restrict__ heat) {
  int img = blockIdx.z;
  int t = blockIdx.x * 256 + threadIdx.x;   // < 147456
  int oy = t / WW, ox = t % WW;
  int po = oy & 1, px = ox & 1;
  int iyb = (oy + po) >> 1, ixb = (ox + px) >> 1;
  int iys[2] = { iyb, iyb - 1 };
  int ixs[2] = { ixb, ixb - 1 };
  int kys[2] = { 1 - po, 3 - po };
  int kxs[2] = { 1 - px, 3 - px };
  float s = b[0];
  for (int i = 0; i < 32; ++i) {
    const _Float16* ip = d2cat + ((size_t)img * 32 + i) * 36864;
    const float* wp = w + i * 16;
#pragma unroll
    for (int ty = 0; ty < 2; ++ty) {
      int iy = iys[ty]; if ((unsigned)iy >= 192u) continue;
#pragma unroll
      for (int txk = 0; txk < 2; ++txk) {
        int ix = ixs[txk]; if ((unsigned)ix >= 192u) continue;
        s += (float)ip[iy * 192 + ix] * wp[kys[ty] * 4 + kxs[txk]];
      }
    }
  }
  heat[(size_t)img * SENT + t] = 1.f / (1.f + expf(-s));
}

// ------------------------- CCL -------------------------------------------
__global__ void ccl_init_k(const float* __restrict__ heat, int* __restrict__ l0) {
  int img = blockIdx.z;
  int t = blockIdx.x * 256 + threadIdx.x;
  l0[(size_t)img * SENT + t] = (heat[(size_t)img * SENT + t] > 0.75f) ? t : SENT;
}

// 32x32 tile with 1-halo, 32 Jacobi min-iterations in LDS per launch.
__global__ void ccl_prop_k(const int* __restrict__ lin, int* __restrict__ lout) {
  const int tid = threadIdx.x;
  const int img = blockIdx.z;
  const size_t base = (size_t)img * SENT;
  const int x0 = blockIdx.x * 32 - 1, y0 = blockIdx.y * 32 - 1;
  __shared__ int S[34][36];
  for (int e = tid; e < 34 * 34; e += 256) {
    int sy = e / 34, sx = e % 34;
    int gy = y0 + sy, gx = x0 + sx;
    int v = SENT;
    if ((unsigned)gy < (unsigned)HH && (unsigned)gx < (unsigned)WW)
      v = lin[base + (size_t)gy * WW + gx];
    S[sy][sx] = v;
  }
  __syncthreads();
  bool fg[4];
#pragma unroll
  for (int k = 0; k < 4; ++k) {
    int idx = tid + k * 256;
    fg[k] = S[(idx >> 5) + 1][(idx & 31) + 1] != SENT;
  }
  for (int it = 0; it < 32; ++it) {
    int m[4];
#pragma unroll
    for (int k = 0; k < 4; ++k) {
      int idx = tid + k * 256, py = (idx >> 5) + 1, pxx = (idx & 31) + 1;
      if (fg[k]) {
        int v = S[py - 1][pxx - 1];
        v = imin2(v, S[py - 1][pxx]);     v = imin2(v, S[py - 1][pxx + 1]);
        v = imin2(v, S[py][pxx - 1]);     v = imin2(v, S[py][pxx]);
        v = imin2(v, S[py][pxx + 1]);
        v = imin2(v, S[py + 1][pxx - 1]); v = imin2(v, S[py + 1][pxx]);
        v = imin2(v, S[py + 1][pxx + 1]);
        m[k] = v;
      } else m[k] = SENT;
    }
    __syncthreads();
#pragma unroll
    for (int k = 0; k < 4; ++k) {
      int idx = tid + k * 256;
      S[(idx >> 5) + 1][(idx & 31) + 1] = m[k];
    }
    __syncthreads();
  }
#pragma unroll
  for (int k = 0; k < 4; ++k) {
    int idx = tid + k * 256, py = idx >> 5, pxx = idx & 31;
    lout[base + (size_t)(blockIdx.y * 32 + py) * WW + blockIdx.x * 32 + pxx] = S[py + 1][pxx + 1];
  }
}

// ------------------------- extract: 16 smallest roots -> polar -------------
__global__ void extract_k(const int* __restrict__ labels, float* __restrict__ rAll,
                          float* __restrict__ aAll, int* __restrict__ nAll, int statBase) {
  int img = blockIdx.x, tid = threadIdx.x;
  const int* L = labels + (size_t)img * SENT;
  __shared__ int red[256];
  __shared__ int u[16];
  __shared__ float scnt[16], ssx[16], ssy[16];
  int last = -1;
  for (int kk = 0; kk < 16; ++kk) {           // k-th smallest root label
    int mn = SENT;
    for (int p = tid; p < SENT; p += 256) {
      int v = L[p];
      if (v == p && v > last && v < mn) mn = v;   // root: label == own index
    }
    red[tid] = mn; __syncthreads();
    for (int s = 128; s > 0; s >>= 1) {
      if (tid < s) red[tid] = imin2(red[tid], red[tid + s]);
      __syncthreads();
    }
    if (tid == 0) u[kk] = red[0];
    __syncthreads();
    last = u[kk];
  }
  if (tid < 16) { scnt[tid] = 0.f; ssx[tid] = 0.f; ssy[tid] = 0.f; }
  __syncthreads();
  int ur[16];
#pragma unroll
  for (int s = 0; s < 16; ++s) ur[s] = u[s];
  float lc[16], lx[16], ly[16];
#pragma unroll
  for (int s = 0; s < 16; ++s) { lc[s] = 0.f; lx[s] = 0.f; ly[s] = 0.f; }
  for (int p = tid; p < SENT; p += 256) {
    int v = L[p];
    if (v < SENT) {
#pragma unroll
      for (int s = 0; s < 16; ++s) {
        if (ur[s] == v) { lc[s] += 1.f; lx[s] += (float)(p % WW); ly[s] += (float)(p / WW); break; }
      }
    }
  }
#pragma unroll
  for (int s = 0; s < 16; ++s) {
    if (lc[s] > 0.f) {
      atomicAdd(&scnt[s], lc[s]); atomicAdd(&ssx[s], lx[s]); atomicAdd(&ssy[s], ly[s]);
    }
  }
  __syncthreads();
  if (tid == 0) {
    float rr[16], aa[16], key[16]; int nv = 0;
    for (int s = 0; s < 16; ++s) {
      bool valid = u[s] < SENT;
      float c = scnt[s] < 1.f ? 1.f : scnt[s];
      float cx = ssx[s] / c, cy = ssy[s] / c;
      float dx = cx - 192.f, dy = cy - 192.f;
      float r = sqrtf(dx * dx + dy * dy);
      float ang = atan2f(dy, dx) * 57.29577951308232f;
      ang = fmodf(ang, 360.f); if (ang < 0.f) ang += 360.f;
      rr[s] = valid ? r : 0.f;
      aa[s] = valid ? ang : 0.f;
      key[s] = valid ? r : __builtin_inff();
      nv += valid ? 1 : 0;
    }
    for (int i = 1; i < 16; ++i) {            // stable insertion sort by radius
      float k = key[i], rv = rr[i], av = aa[i]; int j = i - 1;
      while (j >= 0 && key[j] > k) { key[j+1]=key[j]; rr[j+1]=rr[j]; aa[j+1]=aa[j]; --j; }
      key[j+1] = k; rr[j+1] = rv; aa[j+1] = av;
    }
    int g = statBase + img;
    for (int s = 0; s < 16; ++s) { rAll[g * 16 + s] = rr[s]; aAll[g * 16 + s] = aa[s]; }
    nAll[g] = nv;
  }
}

// ------------------------- greedy match + MLP head -------------------------
__global__ void match_mlp_k(const float* __restrict__ rAll, const float* __restrict__ aAll,
                            const int* __restrict__ nAll,
                            const float* __restrict__ w1, const float* __restrict__ b1,
                            const float* __restrict__ w2, const float* __restrict__ b2,
                            const float* __restrict__ w3, const float* __restrict__ b3,
                            float* __restrict__ out) {
  int b = threadIdx.x; if (b >= 32) return;
  float r1[16], a1[16], r2[16], a2[16];
  for (int i = 0; i < 16; ++i) {
    r1[i] = rAll[b * 16 + i];        a1[i] = aAll[b * 16 + i];
    r2[i] = rAll[(32 + b) * 16 + i]; a2[i] = aAll[(32 + b) * 16 + i];
  }
  int n1 = nAll[b], n2 = nAll[32 + b];
  bool sw = n1 > n2;
  float rA[16], aA[16], rB[16], aB[16];
  for (int i = 0; i < 16; ++i) {
    rA[i] = sw ? r2[i] : r1[i]; aA[i] = sw ? a2[i] : a1[i];
    rB[i] = sw ? r1[i] : r2[i]; aB[i] = sw ? a1[i] : a2[i];
  }
  int nA = sw ? n2 : n1, nB = sw ? n1 : n2;
  bool used[16]; for (int i = 0; i < 16; ++i) used[i] = false;
  float feats[32];
  for (int i = 0; i < 16; ++i) {
    bool vi = i < nA;
    float best = __builtin_inff(); int j = 0;
    for (int jj = 0; jj < 16; ++jj) {
      if (!used[jj] && jj < nB) {
        float d = fabsf(rB[jj] - rA[i]);
        if (d < best) { best = d; j = jj; }   // first-min tie break like argmin
      }
    }
    float dr = rA[i] - rB[j];
    float da = fabsf(sinf((aA[i] - aB[j]) * 0.017453292519943295f));
    feats[2 * i]     = vi ? dr : 0.f;
    feats[2 * i + 1] = vi ? da : 0.f;
    if (vi) used[j] = true;
  }
  float h1[32];
  for (int o = 0; o < 32; ++o) {
    float s = b1[o];
    for (int k = 0; k < 32; ++k) s += feats[k] * w1[o * 32 + k];
    h1[o] = frelu(s);
  }
  float h2[8];
  for (int o = 0; o < 8; ++o) {
    float s = b2[o];
    for (int k = 0; k < 32; ++k) s += h1[k] * w2[o * 32 + k];
    h2[o] = frelu(s);
  }
  float s = b3[0];
  for (int k = 0; k < 8; ++k) s += h2[k] * w3[k];
  out[b] = s;
}

// ---------------------------------------------------------------------------
extern "C" void kernel_launch(void* const* d_in, const int* in_sizes, int n_in,
                              void* d_out, int out_size, void* d_ws, size_t ws_size,
                              hipStream_t stream) {
  (void)in_sizes; (void)n_in; (void)out_size; (void)ws_size;
  const float* x1   = (const float*)d_in[0];
  const float* x2   = (const float*)d_in[1];
  const float* e1_w = (const float*)d_in[2];  const float* e1_b = (const float*)d_in[3];
  const float* e2_w = (const float*)d_in[4];  const float* e2_b = (const float*)d_in[5];
  const float* e3_w = (const float*)d_in[6];  const float* e3_b = (const float*)d_in[7];
  const float* d3_w = (const float*)d_in[8];  const float* d3_b = (const float*)d_in[9];
  const float* d2_w = (const float*)d_in[10]; const float* d2_b = (const float*)d_in[11];
  const float* d1_w = (const float*)d_in[12]; const float* d1_b = (const float*)d_in[13];
  const float* w1   = (const float*)d_in[14]; const float* b1   = (const float*)d_in[15];
  const float* w2   = (const float*)d_in[16]; const float* b2   = (const float*)d_in[17];
  const float* w3   = (const float*)d_in[18]; const float* b3   = (const float*)d_in[19];

  char* ws = (char*)d_ws; size_t off = 0;
  auto alloc = [&](size_t bytes) -> void* {
    void* p = ws + off; off += (bytes + 255) & ~(size_t)255; return p;
  };
  _Float16* e2Bt = (_Float16*)alloc((size_t)32 * 256 * 2);
  _Float16* e3Bt = (_Float16*)alloc((size_t)64 * 512 * 2);
  _Float16* d3Bt = (_Float16*)alloc((size_t)4 * 32 * 256 * 2);
  _Float16* d2Bt = (_Float16*)alloc((size_t)4 * 16 * 256 * 2);
  float* rAll = (float*)alloc(64 * 16 * 4);
  float* aAll = (float*)alloc(64 * 16 * 4);
  int*   nAll = (int*)alloc(64 * 4);
  _Float16* d3cat = (_Float16*)alloc((size_t)IMGS * 64 * 9216 * 2);   // 18.9 MB
  _Float16* d2cat = (_Float16*)alloc((size_t)IMGS * 32 * 36864 * 2);  // 37.7 MB
  _Float16* a3    = (_Float16*)alloc((size_t)IMGS * 64 * 2304 * 2);   //  4.7 MB
  float*    heat  = (float*)alloc((size_t)IMGS * SENT * 4);           //  9.4 MB
  // label ping-pong aliases d2cat (dead once heat is written)
  int* labels0 = (int*)d2cat;
  int* labels1 = labels0 + (size_t)IMGS * SENT;

  pack_conv_k <<<(32 * 256 + 255) / 256, 256, 0, stream>>>(e2_w, e2Bt, 32 * 256);
  pack_conv_k <<<(64 * 512 + 255) / 256, 256, 0, stream>>>(e3_w, e3Bt, 64 * 512);
  pack_convT_k<<<(4 * 32 * 256 + 255) / 256, 256, 0, stream>>>(d3_w, d3Bt, 64, 32);
  pack_convT_k<<<(4 * 16 * 256 + 255) / 256, 256, 0, stream>>>(d2_w, d2Bt, 64, 16);

  for (int chunk = 0; chunk < 4; ++chunk) {
    int side = chunk >> 1, half = chunk & 1;
    const float* x = (side ? x2 : x1) + (size_t)half * IMGS * (HH * WW);
    int statBase = side * 32 + half * IMGS;

    // e1: 1->16 (writes d2cat ch16..31 = skip)
    conv_e1_k<<<dim3(144, 16, IMGS), 256, 0, stream>>>(x, e1_w, e1_b, d2cat);
    // e2: 16->32 (writes d3cat ch32..63 = skip), K=256
    conv_fwd_wmma<<<dim3(6, 96, IMGS), 64, 0, stream>>>(
        d2cat + 16 * 36864, 32 * 36864, 16, 192, 192, e2Bt, 256, e2_b,
        d3cat, 64 * 9216, 32, 32, 96, 96);
    // e3: 32->64 (reads d3cat ch32..63), K=512
    conv_fwd_wmma<<<dim3(3, 48, IMGS), 128, 0, stream>>>(
        d3cat + 32 * 9216, 64 * 9216, 32, 96, 96, e3Bt, 512, e3_b,
        a3, 64 * 2304, 0, 64, 48, 48);
    // d3: convT 64->32 (writes d3cat ch0..31), K=256
    convT_wmma<<<dim3(6, 96, IMGS), 64, 0, stream>>>(
        a3, 64 * 2304, 64, 48, 48, d3Bt, 256, d3_b,
        d3cat, 64 * 9216, 0, 32, 96, 96);
    // d2: convT 64->16 (writes d2cat ch0..15), K=256
    convT_wmma<<<dim3(12, 192, IMGS), 32, 0, stream>>>(
        d3cat, 64 * 9216, 64, 96, 96, d2Bt, 256, d2_b,
        d2cat, 32 * 36864, 0, 16, 192, 192);
    // d1: convT 32->1 + sigmoid
    d1_sigmoid_k<<<dim3(576, 1, IMGS), 256, 0, stream>>>(d2cat, d1_w, d1_b, heat);

    // CCL: init + 16 launches of 32 LDS iterations each
    ccl_init_k<<<dim3(576, 1, IMGS), 256, 0, stream>>>(heat, labels0);
    for (int r = 0; r < 16; ++r)
      ccl_prop_k<<<dim3(12, 12, IMGS), 256, 0, stream>>>(
          (r & 1) ? labels1 : labels0, (r & 1) ? labels0 : labels1);

    extract_k<<<IMGS, 256, 0, stream>>>(labels0, rAll, aAll, nAll, statBase);
  }

  match_mlp_k<<<1, 32, 0, stream>>>(rAll, aAll, nAll, w1, b1, w2, b2, w3, b3,
                                    (float*)d_out);
}
